// GCNEncoder_31542239822439
// MI455X (gfx1250) — compile-verified, matrix operand order checked by reading, not run on hardware
//
#include <hip/hip_runtime.h>
#include <stdint.h>

#define NND   50000
#define NE    800000
#define INCH  512
#define HIDC  416
#define OUTC  208

typedef __attribute__((ext_vector_type(16))) __bf16 v16bf;
typedef __attribute__((ext_vector_type(8)))  __bf16 v8bf;
typedef __attribute__((ext_vector_type(8)))  float  v8f;

// ---- float -> bf16 with round-to-nearest-even (bit trick; no fptrunc dependence) ----
__device__ __forceinline__ __bf16 f2bf(float f) {
  union { float f; uint32_t u; } v; v.f = f;
  uint32_t r = v.u + 0x7FFFu + ((v.u >> 16) & 1u);
  uint16_t h = (uint16_t)(r >> 16);
  __bf16 b;
  __builtin_memcpy(&b, &h, 2);
  return b;
}

// ---------------- small utility kernels ----------------
__global__ void k_fill_f32(float* __restrict__ p, float v, int n) {
  int i = blockIdx.x * blockDim.x + threadIdx.x;
  if (i < n) p[i] = v;
}

__global__ void k_deg_count(const long long* __restrict__ col, float* __restrict__ deg, int e) {
  int i = blockIdx.x * blockDim.x + threadIdx.x;
  if (i < e) atomicAdd(&deg[(int)col[i]], 1.0f);
}

__global__ void k_rsqrt(const float* __restrict__ deg, float* __restrict__ dis, int n) {
  int i = blockIdx.x * blockDim.x + threadIdx.x;
  if (i < n) dis[i] = rsqrtf(deg[i]);   // deg >= 1 always (self loops)
}

__global__ void k_cvt_bf16(const float* __restrict__ in, __bf16* __restrict__ out, int n) {
  int i = blockIdx.x * blockDim.x + threadIdx.x;
  if (i < n) out[i] = f2bf(in[i]);
}

// out[i, f] = bias[f]  (folds the per-node bias into the scatter accumulator init)
__global__ void k_init_bias(float* __restrict__ out, const float* __restrict__ b, int n, int F) {
  int i = blockIdx.x * blockDim.x + threadIdx.x;
  if (i < n) out[i] = b[i % F];
}

// hb = bf16(relu(agg))   (bias already inside agg via k_init_bias)
__global__ void k_relu_cvt(const float* __restrict__ agg, __bf16* __restrict__ hb, int n) {
  int i = blockIdx.x * blockDim.x + threadIdx.x;
  if (i < n) hb[i] = f2bf(fmaxf(agg[i], 0.0f));
}

// ---- pack W [K x N] f32 row-major into WMMA-B fragment layout: [Kt][Nt][32 lanes][16 bf16]
__global__ void k_pack_w(const float* __restrict__ W, __bf16* __restrict__ Wp, int K, int N) {
  int idx = blockIdx.x * blockDim.x + threadIdx.x;
  const int Ntiles = N >> 4;
  const int total = (K >> 5) * Ntiles * 512;
  if (idx >= total) return;
  int j    = idx & 15;
  int lane = (idx >> 4) & 31;
  int nt   = (idx >> 9) % Ntiles;
  int kt   = (idx >> 9) / Ntiles;
  int col  = nt * 16 + (lane & 15);
  int klo  = kt * 32 + ((lane >> 4) ? 8 : 0);
  int k    = klo + (j < 8 ? j : 8 + j);     // j<8 -> K klo+j ; j>=8 -> K klo+16+(j-8)
  Wp[idx] = f2bf(W[(size_t)k * N + col]);
}

// ---------------- WMMA GEMM: C[M,N] = A[M,K](bf16, row major) @ Bp(packed) ----------------
// one wave per 16 x (16*NT) tile; all dims are exact multiples, EXEC stays all-1s.
template <int NT>
__global__ void __launch_bounds__(32)
k_gemm_bf16(const __bf16* __restrict__ A, const __bf16* __restrict__ Bp,
            float* __restrict__ C, int N, int K)
{
  const int lane   = threadIdx.x & 31;
  const int mtile  = blockIdx.x;
  const int ng     = blockIdx.y;
  const int Ntiles = N >> 4;
  const int Kt     = K >> 5;

  const int arow   = mtile * 16 + (lane & 15);
  const int koff   = (lane >> 4) ? 8 : 0;
  const __bf16* ap = A + (size_t)arow * K;

  v8f acc[NT];
#pragma unroll
  for (int t = 0; t < NT; ++t) acc[t] = (v8f){0,0,0,0,0,0,0,0};

  for (int kt = 0; kt < Kt; ++kt) {
    const int kb = kt * 32 + koff;
    v8bf alo = *(const v8bf*)(ap + kb);        // K = kb .. kb+7
    v8bf ahi = *(const v8bf*)(ap + kb + 16);   // K = kb+16 .. kb+23
    v16bf afrag = __builtin_shufflevector(alo, ahi,
        0,1,2,3,4,5,6,7,8,9,10,11,12,13,14,15);
#pragma unroll
    for (int t = 0; t < NT; ++t) {
      const int nt = ng * NT + t;
      v16bf bfrag = *(const v16bf*)(Bp + (((size_t)kt * Ntiles + nt) * 32 + lane) * 16);
      acc[t] = __builtin_amdgcn_wmma_f32_16x16x32_bf16(
          false, afrag, false, bfrag, (short)0, acc[t], false, false);
    }
  }

  // C/D layout: VGPR r -> row (r or r+8), col = lane&15
  const int mrow = mtile * 16 + ((lane >> 4) ? 8 : 0);
  const int c0   = lane & 15;
#pragma unroll
  for (int t = 0; t < NT; ++t) {
    float* cp = C + (size_t)mrow * N + (ng * NT + t) * 16 + c0;
#pragma unroll
    for (int r = 0; r < 8; ++r) cp[(size_t)r * N] = acc[t][r];
  }
}

// ---------------- edge scatter: out[col] += h[row] * dis[row]*dis[col] (+ self loops) ----
__global__ void k_scatter(const float* __restrict__ h,
                          const long long* __restrict__ row,
                          const long long* __restrict__ col,
                          const float* __restrict__ dis,
                          float* __restrict__ out, int E, int nnodes, int F)
{
  const int lane = threadIdx.x & 31;
  const int wid  = (blockIdx.x * blockDim.x + threadIdx.x) >> 5;
  const int nw   = (gridDim.x * blockDim.x) >> 5;
  const int total = E + nnodes;                 // self loops appended
  for (int e = wid; e < total; e += nw) {
    int r, c;
    if (e < E) { r = (int)row[e]; c = (int)col[e]; }
    else       { r = c = e - E; }
    const float w = dis[r] * dis[c];
    const float* hp = h + (size_t)r * F;
    float* op       = out + (size_t)c * F;
    for (int f = lane; f < F; f += 32)
      atomicAdd(op + f, hp[f] * w);
  }
}

// ---------------- driver ----------------
extern "C" void kernel_launch(void* const* d_in, const int* in_sizes, int n_in,
                              void* d_out, int out_size, void* d_ws, size_t ws_size,
                              hipStream_t stream) {
  const float*     x   = (const float*)d_in[0];
  const long long* ei  = (const long long*)d_in[1];
  const float*     W1  = (const float*)d_in[2];
  const float*     b1  = (const float*)d_in[3];
  const float*     W2  = (const float*)d_in[4];
  const float*     b2  = (const float*)d_in[5];
  float*           out = (float*)d_out;

  const long long* erow = ei;
  const long long* ecol = ei + NE;

  // ---- workspace carving (256B aligned) ----
  size_t off = 0;
  char* base = (char*)d_ws;
  auto carve = [&](size_t bytes) -> void* {
    void* p = base + off;
    off += (bytes + 255) & ~(size_t)255;
    return p;
  };
  float*  deg  = (float*)carve((size_t)NND * 4);
  float*  dis  = (float*)carve((size_t)NND * 4);
  __bf16* xb   = (__bf16*)carve((size_t)NND * INCH * 2);
  __bf16* W1p  = (__bf16*)carve((size_t)(INCH/32) * (HIDC/16) * 512 * 2);
  __bf16* W2p  = (__bf16*)carve((size_t)(HIDC/32) * (OUTC/16) * 512 * 2);
  float*  bufA = (float*)carve((size_t)NND * HIDC * 4);   // h1, later reused as bf16 h-relu
  float*  bufB = (float*)carve((size_t)NND * HIDC * 4);   // agg1, later h2
  __bf16* hb   = (__bf16*)bufA;                           // alias: h1 dead after scatter1

  const int T = 256;
  auto G = [&](int n) { return (n + T - 1) / T; };

  // 1) normalization: deg = 1 + in-degree ; dis = rsqrt(deg)
  k_fill_f32 <<<G(NND), T, 0, stream>>>(deg, 1.0f, NND);
  k_deg_count<<<G(NE),  T, 0, stream>>>(ecol, deg, NE);
  k_rsqrt    <<<G(NND), T, 0, stream>>>(deg, dis, NND);

  // 2) bf16 conversions / weight packing
  k_cvt_bf16<<<G(NND * INCH), T, 0, stream>>>(x, xb, NND * INCH);
  k_pack_w  <<<G((INCH/32)*(HIDC/16)*512), T, 0, stream>>>(W1, W1p, INCH, HIDC);
  k_pack_w  <<<G((HIDC/32)*(OUTC/16)*512), T, 0, stream>>>(W2, W2p, HIDC, OUTC);

  // 3) GEMM1: h1 = xb @ W1   (3125 m-tiles, 26 n-tiles -> NT=2, 13 groups)
  k_gemm_bf16<2><<<dim3(NND/16, (HIDC/16)/2), 32, 0, stream>>>(xb, W1p, bufA, HIDC, INCH);

  // 4) aggregate layer 1: agg1 = b1 (broadcast) + scatter(h1 * norm)
  k_init_bias<<<G(NND * HIDC), T, 0, stream>>>(bufB, b1, NND * HIDC, HIDC);
  k_scatter  <<<8192, T, 0, stream>>>(bufA, erow, ecol, dis, bufB, NE, NND, HIDC);

  // 5) relu + bf16: hb = bf16(relu(agg1))   (hb aliases bufA)
  k_relu_cvt<<<G(NND * HIDC), T, 0, stream>>>(bufB, hb, NND * HIDC);

  // 6) GEMM2: h2 = hb @ W2   (13 n-tiles -> NT=1, 13 groups); h2 in bufB (agg1 dead)
  k_gemm_bf16<1><<<dim3(NND/16, OUTC/16), 32, 0, stream>>>(hb, W2p, bufB, OUTC, HIDC);

  // 7) aggregate layer 2 into d_out: out = b2 + scatter(h2 * norm)
  k_init_bias<<<G(NND * OUTC), T, 0, stream>>>(out, b2, NND * OUTC, OUTC);
  k_scatter  <<<8192, T, 0, stream>>>(bufB, erow, ecol, dis, out, NE, NND, OUTC);
}
